// RBFLayer_65481071401981
// MI455X (gfx1250) — matrix operand order repaired and checked by reading, not compiled, not run on hardware
//
#include <hip/hip_runtime.h>
#include <hip/hip_bf16.h>

// ---------------------------------------------------------------------------
// RBF layer: out[b,o] = exp(-(||x_b||^2 + ||c_o||^2 - 2 * <x_b, c_o>))
// B=16384, I=512, O=1024, all fp32.
//
// Cross term via split-bf16 WMMA (hi*hi + hi*lo + lo*hi) for ~fp32 accuracy.
// Fast path (ws large enough): pre-convert x/centers to bf16 hi/lo in d_ws
// once, then the GEMM hot loop stages tiles with GLOBAL_LOAD_ASYNC_TO_LDS_B128
// into double-buffered LDS, pipelined with s_wait_asynccnt 0x8 so the next
// chunk's DMA overlaps the current chunk's 24 WMMAs. Fallback path converts
// in-loop.
// ---------------------------------------------------------------------------

#define RBF_B   16384
#define RBF_I   512
#define RBF_O   1024

typedef __attribute__((ext_vector_type(16))) __bf16 v16bf;
typedef __attribute__((ext_vector_type(8)))  __bf16 v8bf;
typedef __attribute__((ext_vector_type(4)))  __bf16 v4bf;
typedef __attribute__((ext_vector_type(8)))  float  v8f;

static __device__ inline v8f wmma_bf16(v16bf a, v16bf b, v8f c) {
  // (neg_a, A, neg_b, B, c_mod, C, reuse_a, reuse_b)
  return __builtin_amdgcn_wmma_f32_16x16x32_bf16(false, a, false, b,
                                                 (short)0, c, false, false);
}

static __device__ inline v16bf cat8(v8bf lo, v8bf hi) {
  return __builtin_shufflevector(lo, hi, 0, 1, 2, 3, 4, 5, 6, 7,
                                 8, 9, 10, 11, 12, 13, 14, 15);
}

// One 16-byte global->LDS async copy (GVS addressing: saddr base + 32-bit
// byte offset). Tracked by ASYNCcnt.
static __device__ inline void async_copy_b128(unsigned lds_byte_addr,
                                              unsigned global_byte_off,
                                              const void* base) {
  asm volatile("global_load_async_to_lds_b128 %0, %1, %2"
               :
               : "v"(lds_byte_addr), "v"(global_byte_off), "s"(base)
               : "memory");
}

static __device__ inline void wait_asynccnt0() {
  asm volatile("s_wait_asynccnt 0x0" ::: "memory");
}
// Release only the older copy group (async loads complete in order); the
// newest 8 (this wave's just-issued group) may stay in flight.
static __device__ inline void wait_asynccnt_le8() {
  asm volatile("s_wait_asynccnt 0x8" ::: "memory");
}

// LDS byte offset of a shared-memory element (generic LDS address low 32 bits
// == group-segment byte offset).
template <typename T>
static __device__ inline unsigned lds_off(const T* p) {
  return (unsigned)(uintptr_t)p;
}

// ---------------------------------------------------------------------------
// Pre-pass 1: row squared-norms for x (rows 0..16383) and centers
// (16384..17407). One wave32 per row, shuffle-reduce.
// ---------------------------------------------------------------------------
__global__ __launch_bounds__(256) void rbf_rownorm_kernel(
    const float* __restrict__ x, const float* __restrict__ c,
    float* __restrict__ norms) {
  const int tid  = threadIdx.x;
  const int wave = tid >> 5;
  const int lane = tid & 31;                // wave32
  const int row  = blockIdx.x * 8 + wave;   // 0 .. 17407

  const float* __restrict__ src =
      (row < RBF_B) ? (x + (size_t)row * RBF_I)
                    : (c + (size_t)(row - RBF_B) * RBF_I);

  float s = 0.0f;
#pragma unroll
  for (int i = 0; i < 4; ++i) {
    const float4 v = *(const float4*)(src + (size_t)(lane + 32 * i) * 4);
    s += v.x * v.x + v.y * v.y + v.z * v.z + v.w * v.w;
  }
#pragma unroll
  for (int off = 16; off > 0; off >>= 1) s += __shfl_xor(s, off, 32);
  if (lane == 0) norms[row] = s;
}

// ---------------------------------------------------------------------------
// Pre-pass 2 (fast path): split fp32 -> bf16 hi + bf16 lo, stored to d_ws.
// ---------------------------------------------------------------------------
__global__ __launch_bounds__(256) void rbf_convert_kernel(
    const float* __restrict__ x, const float* __restrict__ c,
    __bf16* __restrict__ xh, __bf16* __restrict__ xl,
    __bf16* __restrict__ ch, __bf16* __restrict__ cl) {
  constexpr int XN = RBF_B * RBF_I;             // 8,388,608 floats
  constexpr int CN = RBF_O * RBF_I;             // 524,288 floats
  constexpr int TOTAL4 = (XN + CN) / 4;

  const int stride = gridDim.x * blockDim.x;
  for (int i = blockIdx.x * blockDim.x + threadIdx.x; i < TOTAL4; i += stride) {
    const int e = i * 4;
    const bool isX = (e < XN);
    const float* src = isX ? (x + e) : (c + (e - XN));
    const float4 v = *(const float4*)src;
    const float f[4] = {v.x, v.y, v.z, v.w};
    v4bf hi, lo;
#pragma unroll
    for (int j = 0; j < 4; ++j) {
      const __bf16 h = (__bf16)f[j];
      hi[j] = h;
      lo[j] = (__bf16)(f[j] - (float)h);
    }
    if (isX) {
      *(v4bf*)&xh[e] = hi;
      *(v4bf*)&xl[e] = lo;
    } else {
      *(v4bf*)&ch[e - XN] = hi;
      *(v4bf*)&cl[e - XN] = lo;
    }
  }
}

// ---------------------------------------------------------------------------
// Fast main kernel: 128x128 tile / 256-thread block (8 waves, 4x2 wave grid;
// each wave -> 32x64 = 2x4 fragments). Double-buffered async global->LDS
// staging pipelined against the WMMA chain.
// ---------------------------------------------------------------------------
__global__ __launch_bounds__(256) void rbf_wmma_async_kernel(
    const __bf16* __restrict__ XH, const __bf16* __restrict__ XL,
    const __bf16* __restrict__ CH, const __bf16* __restrict__ CL,
    const float* __restrict__ xsq, const float* __restrict__ csq,
    float* __restrict__ out) {
  constexpr int KC = 32;   // K chunk (= WMMA K)
  constexpr int LP = 40;   // LDS row pitch in bf16 (KC + 8 pad; 80 B, 16B-mult)
  constexpr int TB = 128 * LP;  // one tile buffer, in bf16 elements

  // 4 logical tiles x 2 buffers = 80 KB (WGP has 320 KB).
  __shared__ __bf16 sAh[2 * TB];
  __shared__ __bf16 sAl[2 * TB];
  __shared__ __bf16 sBh[2 * TB];
  __shared__ __bf16 sBl[2 * TB];

  const int tid  = threadIdx.x;
  const int lane = tid & 31;
  const int wave = tid >> 5;
  const int wm   = (wave & 3) * 32;
  const int wn   = (wave >> 2) * 64;

  const int rowBase = blockIdx.y * 128;
  const int colBase = blockIdx.x * 128;

  v8f acc[2][4] = {};

  const int kb  = (lane & 16) ? 8 : 0;
  const int kn0 = (lane & 16) ? 16 : 0;
  const int lr  = lane & 15;

  // Issue one copy group: 4 tiles x 512 chunks of 16 B; 8 async instrs/wave.
  auto stage = [&](int kts, int buf) {
#pragma unroll
    for (int i = 0; i < 2; ++i) {
      const int f   = tid + 256 * i;    // 0..511
      const int r   = f >> 2;           // tile row 0..127
      const int cc8 = (f & 3) * 8;      // bf16 offset within K chunk

      const unsigned goffA =
          (unsigned)(((rowBase + r) * RBF_I + kts + cc8) * 2);
      const unsigned goffB =
          (unsigned)(((colBase + r) * RBF_I + kts + cc8) * 2);
      const unsigned l = (unsigned)((buf * 128 + r) * LP + cc8) * 2u;

      async_copy_b128(lds_off(sAh) + l, goffA, XH);
      async_copy_b128(lds_off(sAl) + l, goffA, XL);
      async_copy_b128(lds_off(sBh) + l, goffB, CH);
      async_copy_b128(lds_off(sBl) + l, goffB, CL);
    }
  };

  stage(0, 0);  // prologue: chunk 0 -> buffer 0

  int buf = 0;
  for (int kt = 0; kt < RBF_I; kt += KC, buf ^= 1) {
    const bool hasNext = (kt + KC) < RBF_I;
    if (hasNext) {
      // Next chunk's DMA overlaps this chunk's WMMAs.
      stage(kt + KC, buf ^ 1);
      wait_asynccnt_le8();   // older group (this chunk's tiles) now resident
    } else {
      wait_asynccnt0();
    }
    __syncthreads();         // all waves' copies for `buf` complete

    // ---- fragments per ISA VGPR layouts --------------------------------
    const int bb = buf * TB;
    v16bf Ah[2], Al[2];
#pragma unroll
    for (int mi = 0; mi < 2; ++mi) {
      const int r = wm + mi * 16 + lr;
      const __bf16* ph = &sAh[bb + r * LP];
      const __bf16* pl = &sAl[bb + r * LP];
      Ah[mi] = cat8(*(const v8bf*)(ph + kb), *(const v8bf*)(ph + 16 + kb));
      Al[mi] = cat8(*(const v8bf*)(pl + kb), *(const v8bf*)(pl + 16 + kb));
    }
    v16bf Bh[4], Bl[4];
#pragma unroll
    for (int ni = 0; ni < 4; ++ni) {
      const int n = wn + ni * 16 + lr;
      const __bf16* ph = &sBh[bb + n * LP];
      const __bf16* pl = &sBl[bb + n * LP];
      Bh[ni] = cat8(*(const v8bf*)(ph + kb), *(const v8bf*)(ph + 16 + kb));
      Bl[ni] = cat8(*(const v8bf*)(pl + kb), *(const v8bf*)(pl + 16 + kb));
    }

    // ---- split-bf16 matmul: hi*hi + hi*lo + lo*hi ----------------------
#pragma unroll
    for (int mi = 0; mi < 2; ++mi) {
#pragma unroll
      for (int ni = 0; ni < 4; ++ni) {
        acc[mi][ni] = wmma_bf16(Ah[mi], Bh[ni], acc[mi][ni]);
        acc[mi][ni] = wmma_bf16(Ah[mi], Bl[ni], acc[mi][ni]);
        acc[mi][ni] = wmma_bf16(Al[mi], Bh[ni], acc[mi][ni]);
      }
    }
    __syncthreads();         // reads of `buf` done before it is re-staged
  }

  // ---- epilogue --------------------------------------------------------
  const int half8 = (lane & 16) ? 8 : 0;
  float xs[2][8];
#pragma unroll
  for (int mi = 0; mi < 2; ++mi)
#pragma unroll
    for (int r = 0; r < 8; ++r)
      xs[mi][r] = xsq[rowBase + wm + mi * 16 + half8 + r];

#pragma unroll
  for (int ni = 0; ni < 4; ++ni) {
    const int col = colBase + wn + ni * 16 + lr;
    const float cs = csq[col];
#pragma unroll
    for (int mi = 0; mi < 2; ++mi) {
#pragma unroll
      for (int r = 0; r < 8; ++r) {
        const float d = xs[mi][r] + cs - 2.0f * acc[mi][ni][r];
        const size_t grow = (size_t)(rowBase + wm + mi * 16 + half8 + r);
        out[grow * RBF_O + col] = __expf(-d);   // GAMMA = 1.0
      }
    }
  }
}

// ---------------------------------------------------------------------------
// Fallback main kernel (small ws): in-loop fp32->bf16 hi/lo staging.
// ---------------------------------------------------------------------------
__global__ __launch_bounds__(256) void rbf_wmma_kernel(
    const float* __restrict__ X, const float* __restrict__ C,
    const float* __restrict__ xsq, const float* __restrict__ csq,
    float* __restrict__ out) {
  constexpr int KC = 32;
  constexpr int LP = 40;

  __shared__ __bf16 sAh[128 * LP];
  __shared__ __bf16 sAl[128 * LP];
  __shared__ __bf16 sBh[128 * LP];
  __shared__ __bf16 sBl[128 * LP];

  const int tid  = threadIdx.x;
  const int lane = tid & 31;
  const int wave = tid >> 5;
  const int wm   = (wave & 3) * 32;
  const int wn   = (wave >> 2) * 64;

  const int rowBase = blockIdx.y * 128;
  const int colBase = blockIdx.x * 128;

  v8f acc[2][4] = {};

  const int kb  = (lane & 16) ? 8 : 0;
  const int kn0 = (lane & 16) ? 16 : 0;
  const int lr  = lane & 15;

  for (int kt = 0; kt < RBF_I; kt += KC) {
#pragma unroll
    for (int i = 0; i < 4; ++i) {
      const int f  = tid + 256 * i;
      const int r  = f >> 3;
      const int cc = (f & 7) * 4;

      const float4 va = *(const float4*)(X + (size_t)(rowBase + r) * RBF_I + kt + cc);
      const float4 vb = *(const float4*)(C + (size_t)(colBase + r) * RBF_I + kt + cc);

      const float fa[4] = {va.x, va.y, va.z, va.w};
      const float fb[4] = {vb.x, vb.y, vb.z, vb.w};
      v4bf ah, al, bh, bl;
#pragma unroll
      for (int j = 0; j < 4; ++j) {
        const __bf16 h0 = (__bf16)fa[j];
        const __bf16 h1 = (__bf16)fb[j];
        ah[j] = h0;
        al[j] = (__bf16)(fa[j] - (float)h0);
        bh[j] = h1;
        bl[j] = (__bf16)(fb[j] - (float)h1);
      }
      *(v4bf*)&sAh[r * LP + cc] = ah;
      *(v4bf*)&sAl[r * LP + cc] = al;
      *(v4bf*)&sBh[r * LP + cc] = bh;
      *(v4bf*)&sBl[r * LP + cc] = bl;
    }
    __syncthreads();

    v16bf Ah[2], Al[2];
#pragma unroll
    for (int mi = 0; mi < 2; ++mi) {
      const int r = wm + mi * 16 + lr;
      const __bf16* ph = &sAh[r * LP];
      const __bf16* pl = &sAl[r * LP];
      Ah[mi] = cat8(*(const v8bf*)(ph + kb), *(const v8bf*)(ph + 16 + kb));
      Al[mi] = cat8(*(const v8bf*)(pl + kb), *(const v8bf*)(pl + 16 + kb));
    }
    v16bf Bh[4], Bl[4];
#pragma unroll
    for (int ni = 0; ni < 4; ++ni) {
      const int n = wn + ni * 16 + lr;
      const __bf16* ph = &sBh[n * LP];
      const __bf16* pl = &sBl[n * LP];
      Bh[ni] = cat8(*(const v8bf*)(ph + kb), *(const v8bf*)(ph + 16 + kb));
      Bl[ni] = cat8(*(const v8bf*)(pl + kb), *(const v8bf*)(pl + 16 + kb));
    }

#pragma unroll
    for (int mi = 0; mi < 2; ++mi) {
#pragma unroll
      for (int ni = 0; ni < 4; ++ni) {
        acc[mi][ni] = wmma_bf16(Ah[mi], Bh[ni], acc[mi][ni]);
        acc[mi][ni] = wmma_bf16(Ah[mi], Bl[ni], acc[mi][ni]);
        acc[mi][ni] = wmma_bf16(Al[mi], Bh[ni], acc[mi][ni]);
      }
    }
    __syncthreads();
  }

  const int half8 = (lane & 16) ? 8 : 0;
  float xs[2][8];
#pragma unroll
  for (int mi = 0; mi < 2; ++mi)
#pragma unroll
    for (int r = 0; r < 8; ++r)
      xs[mi][r] = xsq[rowBase + wm + mi * 16 + half8 + r];

#pragma unroll
  for (int ni = 0; ni < 4; ++ni) {
    const int col = colBase + wn + ni * 16 + lr;
    const float cs = csq[col];
#pragma unroll
    for (int mi = 0; mi < 2; ++mi) {
#pragma unroll
      for (int r = 0; r < 8; ++r) {
        const float d = xs[mi][r] + cs - 2.0f * acc[mi][ni][r];
        const size_t grow = (size_t)(rowBase + wm + mi * 16 + half8 + r);
        out[grow * RBF_O + col] = __expf(-d);
      }
    }
  }
}

// ---------------------------------------------------------------------------
extern "C" void kernel_launch(void* const* d_in, const int* in_sizes, int n_in,
                              void* d_out, int out_size, void* d_ws, size_t ws_size,
                              hipStream_t stream) {
  const float* X = (const float*)d_in[0];   // (16384, 512)
  const float* C = (const float*)d_in[1];   // (1024, 512)
  float* out = (float*)d_out;               // (16384, 1024)

  // ws layout:
  //   [0)            norms: 17408 floats (x then centers)
  //   [128 KB)       xh: 16 MB bf16, xl: 16 MB, ch: 1 MB, cl: 1 MB
  constexpr size_t XN = (size_t)RBF_B * RBF_I;
  constexpr size_t CN = (size_t)RBF_O * RBF_I;
  constexpr size_t CONV_BASE = 131072;  // 128 KB
  constexpr size_t NEED =
      CONV_BASE + 2 * XN * sizeof(__bf16) + 2 * CN * sizeof(__bf16);

  float* norms = (float*)d_ws;
  const float* xsq = norms;
  const float* csq = norms + RBF_B;

  rbf_rownorm_kernel<<<dim3((RBF_B + RBF_O) / 8), dim3(256), 0, stream>>>(
      X, C, norms);

  if (ws_size >= NEED) {
    char* base = (char*)d_ws + CONV_BASE;
    __bf16* xh = (__bf16*)base;
    __bf16* xl = xh + XN;
    __bf16* ch = xl + XN;
    __bf16* cl = ch + CN;

    rbf_convert_kernel<<<dim3(2176), dim3(256), 0, stream>>>(
        X, C, xh, xl, ch, cl);

    rbf_wmma_async_kernel<<<dim3(RBF_O / 128, RBF_B / 128), dim3(256), 0,
                            stream>>>(xh, xl, ch, cl, xsq, csq, out);
  } else {
    rbf_wmma_kernel<<<dim3(RBF_O / 128, RBF_B / 128), dim3(256), 0, stream>>>(
        X, C, xsq, csq, out);
  }
}
